// WindowAttention_gfc_57758720196961
// MI455X (gfx1250) — matrix operand order repaired and checked by reading, not compiled, not run on hardware
//
#include <hip/hip_runtime.h>

// ---------------- problem constants ----------------
#define WS_    7
#define NTOK   49           // tokens per window
#define DIMC   896
#define NHEAD  8
#define HDIM   112
#define NGRP   7
#define CPG_   128
#define NBATCH 2048
#define CN     43904        // DIMC * NTOK (flat qkv size per batch)
#define BIASR  169          // (2*WS-1)^2

// ---------------- CDNA5 WMMA types ----------------
typedef __attribute__((ext_vector_type(16))) __bf16 v16bf;
typedef __attribute__((ext_vector_type(8)))  float  v8f;

union Frag {
  v16bf v;
  uint4 q[2];
  unsigned short s[16];
  __bf16 h[16];
};

// native bf16 convert (lowers to v_cvt_*bf16* on gfx1250, replaces manual RNE bit-twiddle)
__device__ __forceinline__ __bf16 f2bf(float f) { return (__bf16)f; }

__device__ __forceinline__ unsigned short f2bfu(float f) {
  __bf16 h = (__bf16)f;
  return __builtin_bit_cast(unsigned short, h);
}

// shared memory carving (bytes)
#define SM_QKV_BYTES   (3 * CN * 2)                 // 263424: q,k,v bf16, flat (c*49+tok)
#define SM_BIAS_OFF    SM_QKV_BYTES                 // 169*8 f32 = 5408
#define SM_SCR_OFF     (SM_QKV_BYTES + 5408)        // 268832 (16B aligned)
#define SM_TOTAL       (SM_SCR_OFF + 24576)         // 293408 < 320KB WGP LDS

// =====================================================================
// Kernel 1: fused grouped-conv QKV + window attention (one block/window)
// =====================================================================
__global__ __launch_bounds__(256) void fused_win_attn(
    const float* __restrict__ x,          // (B, 896, 49)
    const float* __restrict__ wq,         // (7, 128, 128)
    const float* __restrict__ wk,
    const float* __restrict__ wv,
    const float* __restrict__ bias_table, // (169, 8)
    float* __restrict__ attn_out)         // (B, 896, 49) -> d_out staging
{
  extern __shared__ unsigned char smem[];
  unsigned short* qkv   = (unsigned short*)smem;                  // [3][CN]
  float*          biasS = (float*)(smem + SM_BIAS_OFF);           // [169][8]
  unsigned char*  scr   = smem + SM_SCR_OFF;
  unsigned short* Xt    = (unsigned short*)scr;                   // phase1: [64][128] bf16
  float*          Sm    = (float*)scr;                            // phase2: [64][64] f32
  unsigned short* Pm    = (unsigned short*)(scr + 16384);         // phase2: [64][64] bf16

  const int tid  = threadIdx.x;
  const int wid  = tid >> 5;
  const int lane = tid & 31;
  const int lh   = lane & 15;
  const int kb8  = (lane >> 4) * 8;   // A-matrix K base per lane-half
  const int kb16 = (lane >> 4) * 16;  // unused: B loads also use kb8
  const int b    = blockIdx.x;
  const uint4 z4 = make_uint4(0u, 0u, 0u, 0u);

  for (int i = tid; i < BIASR * NHEAD; i += 256) biasS[i] = bias_table[i];

  const float* xb = x + (size_t)b * CN;

  // ---------------- Phase 1: per-group pointwise QKV (bf16 WMMA) ----------------
  for (int g = 0; g < NGRP; ++g) {
    __syncthreads();                                  // protect Xt reuse
    const float* xg = xb + g * (CPG_ * NTOK);
    if (g + 1 < NGRP) __builtin_prefetch(xb + (g + 1) * (CPG_ * NTOK) + tid, 0, 0);
    // stage X transposed (token-major), tokens padded to 64
    for (int i = tid; i < CPG_ * 64; i += 256) {
      int ic = i >> 6, nt = i & 63;
      Xt[nt * CPG_ + ic] = (nt < NTOK) ? f2bfu(xg[ic * NTOK + nt]) : (unsigned short)0;
    }
    __syncthreads();

    const int mt = wid;                 // wave owns output channels [16*mt, 16*mt+16)
    const int oc = mt * 16 + lh;
    for (int p = 0; p < 3; ++p) {
      const float* wsel = (p == 0) ? wq : (p == 1) ? wk : wv;
      const float* wrow = wsel + ((size_t)g * CPG_ + oc) * CPG_;
      Frag a[4];
      #pragma unroll
      for (int ks = 0; ks < 4; ++ks) {
        int i0 = ks * 32 + kb8;         // elems 0..7 : K = i0..i0+7
        int i1 = i0 + 16;               // elems 8..15: K = i1..i1+7
        float4 r0 = *(const float4*)(wrow + i0);
        float4 r1 = *(const float4*)(wrow + i0 + 4);
        float4 r2 = *(const float4*)(wrow + i1);
        float4 r3 = *(const float4*)(wrow + i1 + 4);
        a[ks].h[0]  = f2bf(r0.x);  a[ks].h[1]  = f2bf(r0.y);
        a[ks].h[2]  = f2bf(r0.z);  a[ks].h[3]  = f2bf(r0.w);
        a[ks].h[4]  = f2bf(r1.x);  a[ks].h[5]  = f2bf(r1.y);
        a[ks].h[6]  = f2bf(r1.z);  a[ks].h[7]  = f2bf(r1.w);
        a[ks].h[8]  = f2bf(r2.x);  a[ks].h[9]  = f2bf(r2.y);
        a[ks].h[10] = f2bf(r2.z);  a[ks].h[11] = f2bf(r2.w);
        a[ks].h[12] = f2bf(r3.x);  a[ks].h[13] = f2bf(r3.y);
        a[ks].h[14] = f2bf(r3.z);  a[ks].h[15] = f2bf(r3.w);
      }
      #pragma unroll
      for (int ntile = 0; ntile < 4; ++ntile) {
        v8f acc = {0.f, 0.f, 0.f, 0.f, 0.f, 0.f, 0.f, 0.f};
        int ntok = ntile * 16 + lh;
        #pragma unroll
        for (int ks = 0; ks < 4; ++ks) {
          Frag bm;
          const uint4* src = (const uint4*)(Xt + ntok * CPG_ + ks * 32 + kb8);
          bm.q[0] = src[0]; bm.q[1] = src[2];
          acc = __builtin_amdgcn_wmma_f32_16x16x32_bf16(
              false, a[ks].v, false, bm.v, (short)0, acc, false, false);
        }
        if (ntok < NTOK) {
          int rbase = mt * 16 + kb8;    // D rows: rbase + r
          unsigned short* dst = qkv + p * CN + (g * CPG_ + rbase) * NTOK + ntok;
          #pragma unroll
          for (int r = 0; r < 8; ++r) dst[r * NTOK] = f2bfu(acc[r]);
        }
      }
    }
  }
  __syncthreads();

  // ---------------- Phase 2: attention per head (all waves cooperate) ----------------
  const unsigned short* Qb = qkv;
  const unsigned short* Kb = qkv + CN;
  const unsigned short* Vb = qkv + 2 * CN;
  const float scale = 0.09449111825230679f;          // 112^-0.5

  for (int h = 0; h < NHEAD; ++h) {
    // ---- S = scale * Q K^T + rel_bias ----
    #pragma unroll
    for (int tt = 0; tt < 2; ++tt) {
      int t = wid * 2 + tt;
      int mt2 = t >> 2, nt2 = t & 3;
      v8f acc = {0.f, 0.f, 0.f, 0.f, 0.f, 0.f, 0.f, 0.f};
      int ntokA = mt2 * 16 + lh;   // query token (A rows)
      int mtokB = nt2 * 16 + lh;   // key token   (B cols)
      #pragma unroll
      for (int ks = 0; ks < 4; ++ks) {
        Frag af, bm;
        if (ntokA < NTOK) {
          const unsigned short* base = Qb + ntokA * DIMC + h * HDIM;
          int d0 = ks * 32 + kb8, d1 = d0 + 16;
          af.q[0] = *(const uint4*)(base + d0);                      // d0+7 <= 111
          af.q[1] = (d1 < HDIM) ? *(const uint4*)(base + d1) : z4;   // pad d>=112
        } else { af.q[0] = z4; af.q[1] = z4; }
        int d0b = ks * 32 + kb8;
        if (mtokB < NTOK) {
          const unsigned short* base = Kb + mtokB * DIMC + h * HDIM + d0b;
          bm.q[0] = *(const uint4*)(base);
          bm.q[1] = (d0b + 16 < HDIM) ? *(const uint4*)(base + 16) : z4;
        } else { bm.q[0] = z4; bm.q[1] = z4; }
        acc = __builtin_amdgcn_wmma_f32_16x16x32_bf16(
            false, af.v, false, bm.v, (short)0, acc, false, false);
      }
      int col = nt2 * 16 + lh;
      int rb  = mt2 * 16 + kb8;
      #pragma unroll
      for (int r = 0; r < 8; ++r) {
        int row = rb + r;
        float sv = -1e30f;                 // masks padded keys in softmax
        if (row < NTOK && col < NTOK) {
          int yi = row / WS_, xi = row % WS_;
          int yj = col / WS_, xj = col % WS_;
          int ridx = (yi - yj + WS_ - 1) * (2 * WS_ - 1) + (xi - xj + WS_ - 1);
          sv = acc[r] * scale + biasS[ridx * NHEAD + h];
        }
        Sm[row * 64 + col] = sv;
      }
    }
    __syncthreads();

    // ---- softmax rows (49 valid; padded cols underflow to 0) ----
    if (tid < 64) {
      float mx = -3.4e38f;
      for (int c2 = 0; c2 < 64; ++c2) mx = fmaxf(mx, Sm[tid * 64 + c2]);
      float sum = 0.f;
      for (int c2 = 0; c2 < 64; ++c2) {
        float e = __expf(Sm[tid * 64 + c2] - mx);
        Sm[tid * 64 + c2] = e; sum += e;
      }
      float inv = 1.f / sum;
      for (int c2 = 0; c2 < 64; ++c2) Pm[tid * 64 + c2] = f2bfu(Sm[tid * 64 + c2] * inv);
    }
    __syncthreads();

    // ---- O = P V ----
    for (int t = wid; t < 28; t += 8) {
      int mt3 = t / 7, nt3 = t % 7;
      v8f acc = {0.f, 0.f, 0.f, 0.f, 0.f, 0.f, 0.f, 0.f};
      int ntokA = mt3 * 16 + lh;   // output token rows
      int dcol  = nt3 * 16 + lh;   // head-dim cols (< 112 always)
      #pragma unroll
      for (int ks = 0; ks < 2; ++ks) {
        Frag af, bm;
        {
          const unsigned short* base = Pm + ntokA * 64 + ks * 32 + kb8;
          af.q[0] = *(const uint4*)(base);
          af.q[1] = *(const uint4*)(base + 16);
        }
        #pragma unroll
        for (int e = 0; e < 16; ++e) {
          int m = ks * 32 + kb8 + (e < 8 ? e : e + 8);  // summed token; guard padded rows (P==0 there)
          bm.s[e] = (m < NTOK) ? Vb[m * DIMC + h * HDIM + dcol] : (unsigned short)0;
        }
        acc = __builtin_amdgcn_wmma_f32_16x16x32_bf16(
            false, af.v, false, bm.v, (short)0, acc, false, false);
      }
      int rb = mt3 * 16 + kb8;
      float* obase = attn_out + (size_t)b * CN + (size_t)(h * HDIM + dcol) * NTOK;
      #pragma unroll
      for (int r = 0; r < 8; ++r) {
        int row = rb + r;
        if (row < NTOK) obase[row] = acc[r];   // (B, C, N) layout
      }
    }
    __syncthreads();
  }
}

// =====================================================================
// Kernel 2: per-channel BN statistics (training-mode batch stats)
// =====================================================================
__global__ __launch_bounds__(256) void bn_stats(const float* __restrict__ attn,
                                                float* __restrict__ stats) {
  const int c = blockIdx.x;
  __shared__ float rs[256], rs2[256];
  float s = 0.f, s2 = 0.f;
  for (int i = threadIdx.x; i < NBATCH * NTOK; i += 256) {
    int bb = i / NTOK, nt = i - bb * NTOK;
    float v = attn[(size_t)bb * CN + c * NTOK + nt];
    s += v; s2 += v * v;
  }
  rs[threadIdx.x] = s; rs2[threadIdx.x] = s2;
  __syncthreads();
  for (int off = 128; off > 0; off >>= 1) {
    if (threadIdx.x < off) {
      rs[threadIdx.x]  += rs[threadIdx.x + off];
      rs2[threadIdx.x] += rs2[threadIdx.x + off];
    }
    __syncthreads();
  }
  if (threadIdx.x == 0) {
    const float invn = 1.f / (float)(NBATCH * NTOK);
    float m = rs[0] * invn;
    stats[c]        = m;
    stats[DIMC + c] = rs2[0] * invn - m * m;   // biased variance
  }
}

// =====================================================================
// Kernel 3: normalize in place with gamma/beta
// =====================================================================
__global__ __launch_bounds__(256) void bn_apply(float* __restrict__ attn,
                                                const float* __restrict__ stats,
                                                const float* __restrict__ gamma,
                                                const float* __restrict__ beta) {
  const size_t total = (size_t)NBATCH * CN;
  for (size_t i = (size_t)blockIdx.x * blockDim.x + threadIdx.x; i < total;
       i += (size_t)gridDim.x * blockDim.x) {
    int c = (int)((i / NTOK) % DIMC);
    float m = stats[c], v = stats[DIMC + c];
    float xv = attn[i];
    attn[i] = (xv - m) * rsqrtf(v + 1e-5f) * gamma[c] + beta[c];
  }
}

extern "C" void kernel_launch(void* const* d_in, const int* in_sizes, int n_in,
                              void* d_out, int out_size, void* d_ws, size_t ws_size,
                              hipStream_t stream) {
  const float* x     = (const float*)d_in[0];
  const float* wq    = (const float*)d_in[1];
  const float* wk    = (const float*)d_in[2];
  const float* wv    = (const float*)d_in[3];
  const float* bias  = (const float*)d_in[4];
  const float* gamma = (const float*)d_in[5];
  const float* beta  = (const float*)d_in[6];
  float* out   = (float*)d_out;
  float* stats = (float*)d_ws;   // 2*896 floats

  hipLaunchKernelGGL(fused_win_attn, dim3(NBATCH), dim3(256), SM_TOTAL, stream,
                     x, wq, wk, wv, bias, out);
  hipLaunchKernelGGL(bn_stats, dim3(DIMC), dim3(256), 0, stream, out, stats);
  hipLaunchKernelGGL(bn_apply, dim3(2048), dim3(256), 0, stream,
                     out, stats, gamma, beta);
}
